// MambaEncoder_5059471475238
// MI455X (gfx1250) — matrix operand order repaired and checked by reading, not compiled, not run on hardware
//
#include <hip/hip_runtime.h>
#include <hip/hip_bf16.h>
#include <cstdint>

typedef __attribute__((ext_vector_type(16))) __bf16        v16bf;
typedef __attribute__((ext_vector_type(8)))  float         v8f;
typedef __attribute__((ext_vector_type(8)))  unsigned int  v8u;

namespace {
constexpr int kB       = 4;
constexpr int kL       = 2048;
constexpr int kDM      = 256;
constexpr int kDI      = 512;
constexpr int kNH      = 8;
constexpr int kHD      = 64;
constexpr int kDXBC    = 768;    // DI + 2*DS
constexpr int kDINP    = 1288;   // 2*DI + 2*DS + NH
constexpr int kDINPP   = 1344;   // padded to multiple of 64 (N-tile of 4 B fragments x 16)
constexpr int kRows    = kB * kL;   // 8192
constexpr float kEPS   = 1e-5f;
constexpr int kT       = 8;      // scan timesteps staged per LDS tile
}

static __device__ __forceinline__ unsigned short f2bf(float f) {
  unsigned u = __float_as_uint(f);
  unsigned r = u + 0x7FFFu + ((u >> 16) & 1u);   // round-to-nearest-even
  return (unsigned short)(r >> 16);
}

static __device__ __forceinline__ float blockReduceSum256(float v) {
  #pragma unroll
  for (int m = 16; m >= 1; m >>= 1) v += __shfl_xor(v, m, 32);
  __shared__ float wsum[8];
  __shared__ float total;
  int lane = threadIdx.x & 31;
  int w    = threadIdx.x >> 5;
  if (lane == 0) wsum[w] = v;
  __syncthreads();
  if (threadIdx.x == 0) {
    float s = 0.f;
    #pragma unroll
    for (int i = 0; i < 8; ++i) s += wsum[i];
    total = s;
  }
  __syncthreads();
  return total;
}

// ---------------- weight prep: fp32 -> bf16, transposed to (N, K) ----------------
__global__ __launch_bounds__(256) void k_prep_wi(const float* __restrict__ Wi,
                                                 unsigned short* __restrict__ Wt) {
  int idx = blockIdx.x * 256 + threadIdx.x;        // over NB * kDINPP * kDM
  int k   = idx % kDM;
  int t   = idx / kDM;
  int n   = t % kDINPP;
  int blk = t / kDINPP;
  if (blk >= 8) return;
  float v = (n < kDINP) ? Wi[((size_t)blk * kDM + k) * kDINP + n] : 0.f;
  Wt[idx] = f2bf(v);                               // layout: blk, n, k
}

__global__ __launch_bounds__(256) void k_prep_wo(const float* __restrict__ Wo,
                                                 unsigned short* __restrict__ Wt) {
  int idx = blockIdx.x * 256 + threadIdx.x;        // over NB * kDM * kDI
  int k   = idx % kDI;
  int t   = idx / kDI;
  int n   = t % kDM;
  int blk = t / kDM;
  if (blk >= 8) return;
  float v = Wo[((size_t)blk * kDI + k) * kDM + n];
  Wt[idx] = f2bf(v);                               // layout: blk, n, k
}

// ---------------- residual plumbing ----------------
__global__ __launch_bounds__(256) void k_copy(const float* __restrict__ src,
                                              float* __restrict__ dst, int flip) {
  int idx = blockIdx.x * 256 + threadIdx.x;        // over kRows*kDM
  int d = idx & 255;
  int t = idx >> 8;
  int l = t & (kL - 1);
  int b = t >> 11;
  int ls = flip ? (kL - 1 - l) : l;
  dst[idx] = src[(((size_t)b * kL + ls) << 8) + d];
}

__global__ __launch_bounds__(256) void k_final_bwd(float* __restrict__ out,
                                                   const float* __restrict__ h,
                                                   const float* __restrict__ res) {
  int idx = blockIdx.x * 256 + threadIdx.x;
  int d = idx & 255;
  int t = idx >> 8;
  int l = t & (kL - 1);
  int b = t >> 11;
  size_t f = (((size_t)b * kL + (kL - 1 - l)) << 8) + d;
  out[idx] += h[f] + res[f];
}

// ---------------- fused RMSNorm (DM=256) + bf16 cast ----------------
__global__ __launch_bounds__(256) void k_rms_cast(const float* __restrict__ res,
                                                  const float* __restrict__ w,
                                                  unsigned short* __restrict__ a) {
  int row = blockIdx.x;
  int c   = threadIdx.x;
  float v  = res[(size_t)row * kDM + c];
  float ss = blockReduceSum256(v * v);
  float sc = rsqrtf(ss * (1.f / kDM) + kEPS);
  a[(size_t)row * kDM + c] = f2bf(v * sc * w[c]);
}

// ---------------- WMMA bf16 GEMM: C(MxN,f32) = A(MxK,bf16) * Bt(NxK,bf16)^T ----
// One wave computes a 16x64 tile (A fragment reused over 4 B fragments);
// 8 waves/block cover 128 rows x 64 cols. K multiple of 32.
// Epilogue modes: 0 = store C; 1 = store C and res += acc; 2 = C = acc + res.
__global__ __launch_bounds__(256) void k_gemm(const unsigned short* __restrict__ A, int lda,
                                              const unsigned short* __restrict__ Bt, int K,
                                              float* __restrict__ C, int ldc, int Nvalid,
                                              float* __restrict__ resp, int mode) {
  int lane = threadIdx.x & 31;
  int wave = threadIdx.x >> 5;
  int m0 = (blockIdx.x * 8 + wave) * 16;
  int n0 = blockIdx.y * 64;
  int lr  = lane & 15;
  int hi  = lane >> 4;
  int kbA = hi * 16;   // A: lanes 0-15 hold K 0-15; lanes 16-31 hold K 16-31
  int kbB = hi * 16;   // B: lanes 0-15 hold K 0-15; lanes 16-31 hold K 16-31

  const unsigned short* arow = A  + (size_t)(m0 + lr) * lda;
  const unsigned short* bcol = Bt + (size_t)(n0 + lr) * K;
  const size_t bstep = (size_t)16 * K;   // next 16-col fragment

  v8f acc0 = {}, acc1 = {}, acc2 = {}, acc3 = {};
  for (int k0 = 0; k0 < K; k0 += 32) {
    v8u au, b0, b1, b2, b3;
    #pragma unroll
    for (int j = 0; j < 8; ++j) {
      au[j]     = *(const unsigned*)(arow + k0 + kbA + 2 * j);
    }
    #pragma unroll
    for (int j = 0; j < 8; ++j) {
      b0[j] = *(const unsigned*)(bcol + 0 * bstep + k0 + kbB + 2 * j);
      b1[j] = *(const unsigned*)(bcol + 1 * bstep + k0 + kbB + 2 * j);
      b2[j] = *(const unsigned*)(bcol + 2 * bstep + k0 + kbB + 2 * j);
      b3[j] = *(const unsigned*)(bcol + 3 * bstep + k0 + kbB + 2 * j);
    }
    v16bf av = __builtin_bit_cast(v16bf, au);
    acc0 = __builtin_amdgcn_wmma_f32_16x16x32_bf16(false, av, false,
             __builtin_bit_cast(v16bf, b0), (short)0, acc0, false, false);
    acc1 = __builtin_amdgcn_wmma_f32_16x16x32_bf16(false, av, false,
             __builtin_bit_cast(v16bf, b1), (short)0, acc1, false, false);
    acc2 = __builtin_amdgcn_wmma_f32_16x16x32_bf16(false, av, false,
             __builtin_bit_cast(v16bf, b2), (short)0, acc2, false, false);
    acc3 = __builtin_amdgcn_wmma_f32_16x16x32_bf16(false, av, false,
             __builtin_bit_cast(v16bf, b3), (short)0, acc3, false, false);
  }

  size_t mrow = (size_t)(m0 + 8 * hi);
  #pragma unroll
  for (int t = 0; t < 4; ++t) {
    v8f acc = (t == 0) ? acc0 : (t == 1) ? acc1 : (t == 2) ? acc2 : acc3;
    int n = n0 + t * 16 + lr;
    if (n >= Nvalid) continue;
    float*       crow = C + mrow * ldc + n;
    float*       rrow = resp ? (resp + mrow * ldc + n) : nullptr;
    #pragma unroll
    for (int v = 0; v < 8; ++v) {
      float val = acc[v];
      if (mode == 2)      crow[(size_t)v * ldc] = val + rrow[(size_t)v * ldc];
      else {
        crow[(size_t)v * ldc] = val;
        if (mode == 1) rrow[(size_t)v * ldc] += val;
      }
    }
  }
}

// ---------------- causal depthwise conv (K=4) + SiLU ----------------
__global__ __launch_bounds__(256) void k_convsilu(const float* __restrict__ zx,
                                                  const float* __restrict__ cw,
                                                  const float* __restrict__ cb,
                                                  float* __restrict__ xbc) {
  int idx = blockIdx.x * 256 + threadIdx.x;        // over kRows*kDXBC
  int c   = idx % kDXBC;
  int row = idx / kDXBC;
  int l   = row & (kL - 1);
  float acc = cb[c];
  #pragma unroll
  for (int k = 0; k < 4; ++k) {
    int ls = l - 3 + k;
    if (ls >= 0)
      acc = fmaf(cw[c * 4 + k], zx[(size_t)(row - 3 + k) * kDINPP + kDI + c], acc);
  }
  xbc[idx] = acc / (1.f + __expf(-acc));           // silu
}

// ---------------- dt = softplus(raw + bias) ----------------
__global__ __launch_bounds__(256) void k_dt(const float* __restrict__ zx,
                                            const float* __restrict__ dtb,
                                            float* __restrict__ dt) {
  int idx = blockIdx.x * 256 + threadIdx.x;        // over kRows*kNH
  int hh  = idx & 7;
  int row = idx >> 3;
  float v = zx[(size_t)row * kDINPP + (kDI + kDXBC) + hh] + dtb[hh];
  dt[idx] = (v > 20.f) ? v : log1pf(__expf(v));
}

// ---------------- SSD scan: one workgroup per (b,h); state in registers -------
// 8 timesteps staged per LDS tile; next tile prefetched during compute.
__global__ __launch_bounds__(256) void k_scan(const float* __restrict__ xbc,
                                              const float* __restrict__ dtbuf,
                                              const float* __restrict__ Al,
                                              const float* __restrict__ Dp,
                                              float* __restrict__ ybuf) {
  __shared__ float stage[kT * 320];   // per step: [0..63]=x, [64..191]=B, [192..319]=C
  __shared__ float sdt[kT];
  int b = blockIdx.x >> 3;
  int h = blockIdx.x & 7;
  const float A  = -__expf(Al[h]);
  const float Dv = Dp[h];
  int tid = threadIdx.x;
  int p   = tid >> 2;          // 0..63   (head dim)
  int q   = tid & 3;           // 4 lanes share p, each owns 32 n's
  int nb  = q * 32;
  float hs[32];
  #pragma unroll
  for (int j = 0; j < 32; ++j) hs[j] = 0.f;
  size_t rowbase = (size_t)b * kL;

  for (int t0 = 0; t0 < kL; t0 += kT) {
    // cooperative tile load: kT * 320 floats
    #pragma unroll
    for (int j = 0; j < (kT * 320) / 256; ++j) {
      int e = tid + j * 256;
      int s = e / 320;
      int c = e - s * 320;
      int col = (c < 64) ? (h * kHD + c) : (kDI + (c - 64));
      stage[e] = xbc[(rowbase + t0 + s) * kDXBC + col];
    }
    if (tid < kT) sdt[tid] = dtbuf[(rowbase + t0 + tid) * kNH + h];
    __syncthreads();

    // prefetch next tile's cachelines (global_prefetch_b8) while computing
    if (t0 + kT < kL && tid < kT * 10) {
      int s  = tid / 10;
      int li = tid - s * 10;                           // 10 x 32-float lines per row
      int col = (li < 2) ? (h * kHD + li * 32) : (kDI + (li - 2) * 32);
      __builtin_prefetch(&xbc[(rowbase + t0 + kT + s) * kDXBC + col], 0, 1);
    }

    #pragma unroll
    for (int s = 0; s < kT; ++s) {
      float dtv = sdt[s];
      float dec = __expf(dtv * A);
      float xp  = stage[s * 320 + p];
      float u   = dtv * xp;
      const float* bc = &stage[s * 320 + 64];
      float part = 0.f;
      #pragma unroll
      for (int j = 0; j < 32; ++j) {
        hs[j] = fmaf(hs[j], dec, u * bc[nb + j]);
        part  = fmaf(hs[j], bc[128 + nb + j], part);
      }
      part += __shfl_xor(part, 1, 32);
      part += __shfl_xor(part, 2, 32);
      if (q == 0) ybuf[(rowbase + t0 + s) * kDI + h * kHD + p] = fmaf(Dv, xp, part);
    }
    __syncthreads();
  }
}

// ---------------- gated RMSNorm (DI=512) + bf16 cast ----------------
__global__ __launch_bounds__(256) void k_gnorm(const float* __restrict__ ybuf,
                                               const float* __restrict__ zx,
                                               const float* __restrict__ gw,
                                               unsigned short* __restrict__ g) {
  int row = blockIdx.x;
  int c0 = threadIdx.x;
  int c1 = threadIdx.x + 256;
  float z0 = zx[(size_t)row * kDINPP + c0];
  float z1 = zx[(size_t)row * kDINPP + c1];
  float y0 = ybuf[(size_t)row * kDI + c0];
  float y1 = ybuf[(size_t)row * kDI + c1];
  float v0 = y0 * (z0 / (1.f + __expf(-z0)));
  float v1 = y1 * (z1 / (1.f + __expf(-z1)));
  float ss = blockReduceSum256(v0 * v0 + v1 * v1);
  float sc = rsqrtf(ss * (1.f / kDI) + kEPS);
  g[(size_t)row * kDI + c0] = f2bf(v0 * sc * gw[c0]);
  g[(size_t)row * kDI + c1] = f2bf(v1 * sc * gw[c1]);
}

// ---------------- host orchestration ----------------
extern "C" void kernel_launch(void* const* d_in, const int* in_sizes, int n_in,
                              void* d_out, int out_size, void* d_ws, size_t ws_size,
                              hipStream_t stream) {
  (void)in_sizes; (void)n_in; (void)out_size; (void)ws_size;
  const float* x   = (const float*)d_in[0];
  const float* nw  = (const float*)d_in[1];
  const float* Wi  = (const float*)d_in[2];
  const float* cw  = (const float*)d_in[3];
  const float* cb  = (const float*)d_in[4];
  const float* dtb = (const float*)d_in[5];
  const float* Al  = (const float*)d_in[6];
  const float* Dp  = (const float*)d_in[7];
  const float* gw  = (const float*)d_in[8];
  const float* Wo  = (const float*)d_in[9];
  float* out = (float*)d_out;

  char* ws = (char*)d_ws;
  size_t off = 0;
  auto alloc = [&](size_t bytes) {
    void* p = ws + off;
    off += (bytes + 255) & ~(size_t)255;
    return p;
  };
  float*          res   = (float*)alloc((size_t)kRows * kDM * 4);
  float*          hbuf  = (float*)alloc((size_t)kRows * kDM * 4);
  float*          zx    = (float*)alloc((size_t)kRows * kDINPP * 4);
  float*          xbc   = (float*)alloc((size_t)kRows * kDXBC * 4);
  float*          dtbuf = (float*)alloc((size_t)kRows * kNH * 4);
  float*          ybuf  = (float*)alloc((size_t)kRows * kDI * 4);
  unsigned short* abf   = (unsigned short*)alloc((size_t)kRows * kDI * 2);
  unsigned short* wit   = (unsigned short*)alloc((size_t)8 * kDINPP * kDM * 2);
  unsigned short* wot   = (unsigned short*)alloc((size_t)8 * kDM * kDI * 2);

  dim3 blk(256);
  k_prep_wi<<<(8 * kDINPP * kDM) / 256, blk, 0, stream>>>(Wi, wit);
  k_prep_wo<<<(8 * kDM * kDI) / 256, blk, 0, stream>>>(Wo, wot);

  for (int dir = 0; dir < 2; ++dir) {
    int blk0 = dir * 4;
    k_copy<<<kRows * kDM / 256, blk, 0, stream>>>(x, res, dir);
    for (int i = 0; i < 4; ++i) {
      int bi = blk0 + i;
      k_rms_cast<<<kRows, blk, 0, stream>>>(res, nw + (size_t)bi * kDM, abf);
      // in_proj: M=8192, K=256, N=1288 (padded 1344)
      k_gemm<<<dim3(kRows / 128, kDINPP / 64), blk, 0, stream>>>(
          abf, kDM, wit + (size_t)bi * kDINPP * kDM, kDM, zx, kDINPP, kDINP,
          nullptr, 0);
      k_convsilu<<<kRows * kDXBC / 256, blk, 0, stream>>>(
          zx, cw + (size_t)bi * kDXBC * 4, cb + (size_t)bi * kDXBC, xbc);
      k_dt<<<kRows * kNH / 256, blk, 0, stream>>>(zx, dtb + (size_t)bi * kNH, dtbuf);
      k_scan<<<kB * kNH, blk, 0, stream>>>(xbc, dtbuf, Al + (size_t)bi * kNH,
                                           Dp + (size_t)bi * kNH, ybuf);
      k_gnorm<<<kRows, blk, 0, stream>>>(ybuf, zx, gw + (size_t)bi * kDI, abf);
      // out_proj: M=8192, K=512, N=256. Fused epilogue:
      //   i<3           : h -> hbuf, res += h
      //   i==3 (fwd dir): out = h + res        (final forward result)
      //   i==3 (bwd dir): h -> hbuf            (flip-add done by k_final_bwd)
      if (i < 3) {
        k_gemm<<<dim3(kRows / 128, kDM / 64), blk, 0, stream>>>(
            abf, kDI, wot + (size_t)bi * kDM * kDI, kDI, hbuf, kDM, kDM, res, 1);
      } else if (dir == 0) {
        k_gemm<<<dim3(kRows / 128, kDM / 64), blk, 0, stream>>>(
            abf, kDI, wot + (size_t)bi * kDM * kDI, kDI, out, kDM, kDM, res, 2);
      } else {
        k_gemm<<<dim3(kRows / 128, kDM / 64), blk, 0, stream>>>(
            abf, kDI, wot + (size_t)bi * kDM * kDI, kDI, hbuf, kDM, kDM, nullptr, 0);
      }
    }
    if (dir == 1)
      k_final_bwd<<<kRows * kDM / 256, blk, 0, stream>>>(out, hbuf, res);
  }
}
